// NeuralSplineNetwork_70566312673281
// MI455X (gfx1250) — matrix operand order repaired and checked by reading, not compiled, not run on hardware
//
#include <hip/hip_runtime.h>
#include <hip/hip_bf16.h>
#include <math.h>

#define B_N   8192
#define D_N   256
#define H_N   512
#define NB_N  8
#define PM_N  23           // 3*NB-1
#define NOUT  (D_N * PM_N) // 5888
#define TAILF 3.0f
#define MINWF 0.001f
#define MINHF 0.001f
#define MINDF 0.001f

typedef __attribute__((ext_vector_type(16))) _Float16 v16h;
typedef __attribute__((ext_vector_type(8)))  _Float16 v8h;
typedef __attribute__((ext_vector_type(8)))  float    v8f;

// ---------------------------------------------------------------------------
// Fragment loaders matching CDNA5 wave32 WMMA 16-bit layouts (ISA 7.12.2).
// A: 16x32 (MxK) row-major, stride ld.  lane = row(0..15) + khalf*16.
// ---------------------------------------------------------------------------
__device__ __forceinline__ v16h frag_a(const _Float16* __restrict__ p, int ld, int lane) {
  int row = lane & 15;
  int kb  = (lane >> 4) << 3;              // 0 or 8
  const _Float16* q = p + (size_t)row * ld + kb;
  v8h lo = *(const v8h*)q;                  // K = kb .. kb+7     -> VGPR 0-3
  v8h hi = *(const v8h*)(q + 16);           // K = kb+16 .. kb+23 -> VGPR 4-7
  v16h r;
#pragma unroll
  for (int i = 0; i < 8; ++i) { r[i] = lo[i]; r[i + 8] = hi[i]; }
  return r;
}

// B: 32x16 (KxN); stored transposed [N][K] so each lane's column is contiguous:
// lane holds col = lane&15, K = kb + 0..7 and kb + 16..23, kb = (lane>=16?8:0)
__device__ __forceinline__ v16h frag_b(const _Float16* __restrict__ p, int ld, int lane) {
  int col = lane & 15;
  int kb  = (lane >> 4) << 3;              // 0 or 8
  const _Float16* q = p + (size_t)col * ld + kb;
  v8h lo = *(const v8h*)q;                  // K = kb    .. kb+7  -> VGPR 0-3
  v8h hi = *(const v8h*)(q + 16);           // K = kb+16 .. kb+23 -> VGPR 4-7
  v16h r;
#pragma unroll
  for (int i = 0; i < 8; ++i) { r[i] = lo[i]; r[i + 8] = hi[i]; }
  return r;
}

// ---------------------------------------------------------------------------
// Weight prep: apply autoregressive mask (from indices) + fp32[K][N] -> f16[N][K].
// mode 0 (M0): k <= n%255 ; mode 1 (MH): k%255 <= n%255 ; mode 2 (MF): k%255 < n/23
// ---------------------------------------------------------------------------
__global__ void wconv_t(const float* __restrict__ W, _Float16* __restrict__ Wt,
                        int K, int N, int mode) {
  int total = K * N;
  for (int idx = blockIdx.x * blockDim.x + threadIdx.x; idx < total;
       idx += gridDim.x * blockDim.x) {
    int n = idx / K;
    int k = idx - n * K;
    bool keep;
    if (mode == 0)      keep = (k <= (n % 255));
    else if (mode == 1) keep = ((k % 255) <= (n % 255));
    else                keep = ((k % 255) < (n / PM_N));
    float v = keep ? W[(size_t)k * N + n] : 0.0f;
    Wt[idx] = (_Float16)v;
  }
}

__global__ void cvt16(const float* __restrict__ src, _Float16* __restrict__ dst, int n) {
  for (int i = blockIdx.x * blockDim.x + threadIdx.x; i < n;
       i += gridDim.x * blockDim.x)
    dst[i] = (_Float16)src[i];
}

// ---------------------------------------------------------------------------
// WMMA GEMM with fused epilogue (all variants compile-time specialized):
//   v = A16[M,K] * Bt16[N,K]^T + bias[N]  (+ skip[M,N] if HAS_SKIP)
//   if OUT32: C32 = v        if OUT16: C16 = f16(RELU16 ? relu(v) : v)
// block = 256 threads (8 waves), block tile 128x128, wave tile 32x64.
// ---------------------------------------------------------------------------
template <bool HAS_SKIP, bool OUT32, bool OUT16, bool RELU16>
__global__ __launch_bounds__(256) void gemm_wmma(
    const _Float16* __restrict__ A, const _Float16* __restrict__ Bt,
    const float* __restrict__ bias, const float* __restrict__ skip,
    float* __restrict__ C32, _Float16* __restrict__ C16, int M, int N, int K) {
  int lane = threadIdx.x & 31;
  int wave = threadIdx.x >> 5;
  int m0 = blockIdx.x * 128 + (wave >> 1) * 32;
  int n0 = blockIdx.y * 128 + (wave & 1) * 64;

  v8f acc[2][4];
#pragma unroll
  for (int i = 0; i < 2; ++i)
#pragma unroll
    for (int j = 0; j < 4; ++j)
#pragma unroll
      for (int r = 0; r < 8; ++r) acc[i][j][r] = 0.0f;

  for (int k0 = 0; k0 < K; k0 += 32) {
    v16h a[2], b[4];
#pragma unroll
    for (int i = 0; i < 2; ++i)
      a[i] = frag_a(A + (size_t)(m0 + 16 * i) * K + k0, K, lane);
#pragma unroll
    for (int j = 0; j < 4; ++j)
      b[j] = frag_b(Bt + (size_t)(n0 + 16 * j) * K + k0, K, lane);
#pragma unroll
    for (int i = 0; i < 2; ++i)
#pragma unroll
      for (int j = 0; j < 4; ++j)
        acc[i][j] = __builtin_amdgcn_wmma_f32_16x16x32_f16(
            false, a[i], false, b[j], (short)0, acc[i][j], false, false);
  }

  int cn  = lane & 15;
  int rb0 = (lane >> 4) * 8;   // C/D layout: VGPR r holds M = r + 8*(lane>=16)
#pragma unroll
  for (int i = 0; i < 2; ++i)
#pragma unroll
    for (int j = 0; j < 4; ++j) {
      int n = n0 + 16 * j + cn;
      float bv = bias[n];
#pragma unroll
      for (int r = 0; r < 8; ++r) {
        int m   = m0 + 16 * i + rb0 + r;
        int idx = m * N + n;
        float v = acc[i][j][r] + bv;
        if constexpr (HAS_SKIP) v += skip[idx];
        if constexpr (OUT32) C32[idx] = v;
        if constexpr (OUT16) {
          float o = RELU16 ? fmaxf(v, 0.0f) : v;
          C16[idx] = (_Float16)o;
        }
      }
    }
}

__device__ __forceinline__ float softplusf(float v) {
  return (v > 0.0f) ? (v + log1pf(__expf(-v))) : log1pf(__expf(v));
}

// ---------------------------------------------------------------------------
// Fused final GEMM + RQ spline. grid = (B/16, 16). Each WG: 16 rows x 368 cols
// (16 dims x 23 params) in LDS (stride 372 splits half-wave banks).
// Every wave owns exactly 3 column tiles (t = min(wave*3+u, 22); wave 7's third
// tile duplicates tile 22 -> identical results, benign duplicate LDS write) so
// all accumulators stay in VGPRs (no runtime-indexed arrays -> no scratch).
// Spline bin lookup is a branch-free select-scan: no dynamically indexed knot
// arrays. Emits z (f32), z16 (next flow's A) and deterministic logdet partials.
// ---------------------------------------------------------------------------
#define PSTRIDE 372
__global__ __launch_bounds__(256) void gemm_spline(
    const _Float16* __restrict__ H16,   // [B, 512]
    const _Float16* __restrict__ Wft,   // [5888, 512] transposed masked f16
    const float* __restrict__ bf,       // [5888]
    float* __restrict__ z,              // [B, 256] in/out
    _Float16* __restrict__ z16,         // [B, 256] out
    float* __restrict__ ldp,            // [B, 16] logdet partials
    int accumulate) {
  __shared__ float P[16][PSTRIDE];

  int rbk  = blockIdx.x;          // 16-row block
  int g    = blockIdx.y;          // dim group: dims g*16 .. g*16+15
  int lane = threadIdx.x & 31;
  int wave = threadIdx.x >> 5;
  const _Float16* Ab = H16 + (size_t)rbk * 16 * H_N;
  int colbase = g * 368;

  int tt[3];
#pragma unroll
  for (int u = 0; u < 3; ++u) {
    int t = wave * 3 + u;
    tt[u] = (t > 22) ? 22 : t;     // compile-time unrolled, affine per u
  }

  v8f acc[3];
#pragma unroll
  for (int u = 0; u < 3; ++u)
#pragma unroll
    for (int r = 0; r < 8; ++r) acc[u][r] = 0.0f;

  for (int k0 = 0; k0 < H_N; k0 += 32) {
    v16h a = frag_a(Ab + k0, H_N, lane);
#pragma unroll
    for (int u = 0; u < 3; ++u) {
      v16h b = frag_b(Wft + (size_t)(colbase + tt[u] * 16) * H_N + k0, H_N, lane);
      acc[u] = __builtin_amdgcn_wmma_f32_16x16x32_f16(
          false, a, false, b, (short)0, acc[u], false, false);
    }
  }

  int cn  = lane & 15;
  int rb0 = (lane >> 4) * 8;
#pragma unroll
  for (int u = 0; u < 3; ++u) {
    int ct = tt[u] * 16 + cn;
    float bv = bf[colbase + ct];
#pragma unroll
    for (int r = 0; r < 8; ++r)
      P[rb0 + r][ct] = acc[u][r] + bv;   // wave7 u=2 duplicates u=1: same data
  }
  __syncthreads();

  // ------------------- spline: one (row, dim) per thread -------------------
  int row = threadIdx.x >> 4;     // 0..15
  int dl  = threadIdx.x & 15;     // 0..15
  int b   = rbk * 16 + row;
  int d   = g * 16 + dl;
  const float* pp = &P[row][dl * PM_N];

  const float inv_scale = 0.044194173824159216f;  // 1/sqrt(512)
  float x  = z[b * D_N + d];
  float xc = fminf(fmaxf(x, -TAILF), TAILF);

  // softmax numerators for widths / heights (fully unrolled -> registers)
  float uw[NB_N], uh[NB_N];
#pragma unroll
  for (int j = 0; j < NB_N; ++j) { uw[j] = pp[j] * inv_scale; uh[j] = pp[NB_N + j] * inv_scale; }
  float mw = uw[0], mh = uh[0];
#pragma unroll
  for (int j = 1; j < NB_N; ++j) { mw = fmaxf(mw, uw[j]); mh = fmaxf(mh, uh[j]); }
  float sw = 0.0f, sh = 0.0f;
  float ew[NB_N], eh[NB_N];
#pragma unroll
  for (int j = 0; j < NB_N; ++j) {
    ew[j] = __expf(uw[j] - mw); sw += ew[j];
    eh[j] = __expf(uh[j] - mh); sh += eh[j];
  }
  float rw = (1.0f - MINWF * NB_N) / sw;
  float rh = (1.0f - MINHF * NB_N) / sh;

  // Branch-free bin select-scan. Bin j is latched whenever xc >= cw[j]
  // (cw[0] = -TAIL always matches, later bins overwrite); equals the
  // reference's clip(sum(xc >= cw)-1, 0, NB-1).
  float csw = 0.0f, csh = 0.0f;
  float cwj = -TAILF, chj = -TAILF;     // left knots of bin j
  float dj  = 1.0f;                     // dd[0] = 1 (pad term collapses)
  float icw = -TAILF, ibw = 1.0f, ich = -TAILF, ih = 1.0f, d0 = 1.0f, d1 = 1.0f;
#pragma unroll
  for (int j = 0; j < NB_N; ++j) {
    csw += MINWF + rw * ew[j];
    csh += MINHF + rh * eh[j];
    float cwn = (j == NB_N - 1) ? TAILF : (2.0f * TAILF * csw - TAILF);
    float chn = (j == NB_N - 1) ? TAILF : (2.0f * TAILF * csh - TAILF);
    float dn  = (j == NB_N - 1) ? 1.0f  : (MINDF + softplusf(pp[2 * NB_N + j]));
    bool sel = (xc >= cwj);             // always true at j==0
    if (sel) {
      icw = cwj; ibw = cwn - cwj;
      ich = chj; ih  = chn - chj;
      d0  = dj;  d1  = dn;
    }
    cwj = cwn; chj = chn; dj = dn;
  }

  float delta = ih / ibw;
  float theta = (xc - icw) / ibw;
  float t1m   = theta * (1.0f - theta);
  float denom = delta + (d0 + d1 - 2.0f * delta) * t1m;
  float num   = ih * (delta * theta * theta + d0 * t1m);
  float y     = ich + num / denom;
  float om    = 1.0f - theta;
  float dnum  = delta * delta * (d1 * theta * theta + 2.0f * delta * t1m + d0 * om * om);
  float ldv   = __logf(dnum) - 2.0f * __logf(denom);

  bool inside = (x >= -TAILF) && (x <= TAILF);
  float zn = inside ? y : x;
  z[b * D_N + d]   = zn;
  z16[b * D_N + d] = (_Float16)zn;
  float s = inside ? ldv : 0.0f;

  // deterministic reduce over the 16 dims of this group (one 16-lane cluster)
  for (int off = 8; off >= 1; off >>= 1) s += __shfl_xor(s, off, 16);
  if (dl == 0) {
    if (accumulate) ldp[b * 16 + g] += s;
    else            ldp[b * 16 + g]  = s;
  }
}

// one block per batch row: coalesced z^2 + ldp reduction
__global__ __launch_bounds__(256) void finalize_kernel(
    const float* __restrict__ z, const float* __restrict__ ldp,
    float* __restrict__ out) {
  __shared__ float red[8];
  int b = blockIdx.x;
  float v = z[b * D_N + threadIdx.x];
  float s = v * v;
#pragma unroll
  for (int off = 16; off >= 1; off >>= 1) s += __shfl_xor(s, off, 32);
  int wave = threadIdx.x >> 5;
  int lane = threadIdx.x & 31;
  if (lane == 0) red[wave] = s;
  __syncthreads();
  if (threadIdx.x == 0) {
    float tot = 0.0f;
#pragma unroll
    for (int w = 0; w < 8; ++w) tot += red[w];
    float l = 0.0f;
#pragma unroll
    for (int g = 0; g < 16; ++g) l += ldp[b * 16 + g];
    out[b] = l - 128.0f * logf(6.283185307179586f) - 0.5f * tot;
  }
}

// ---------------------------------------------------------------------------
extern "C" void kernel_launch(void* const* d_in, const int* in_sizes, int n_in,
                              void* d_out, int out_size, void* d_ws, size_t ws_size,
                              hipStream_t stream) {
  const float* x  = (const float*)d_in[0];
  const float* W0 = (const float*)d_in[1];
  const float* b0 = (const float*)d_in[2];
  const float* Wa = (const float*)d_in[3];
  const float* ba = (const float*)d_in[4];
  const float* Wb = (const float*)d_in[5];
  const float* bb = (const float*)d_in[6];
  const float* Wf = (const float*)d_in[7];
  const float* bf = (const float*)d_in[8];

  char* ws = (char*)d_ws;
  size_t off = 0;
  auto take = [&](size_t bytes) -> void* {
    void* p = ws + off;
    off = (off + bytes + 255) & ~(size_t)255;
    return p;
  };
  _Float16* W0t = (_Float16*)take(4ULL * H_N * D_N * 2);
  _Float16* Wat = (_Float16*)take(8ULL * H_N * H_N * 2);
  _Float16* Wbt = (_Float16*)take(8ULL * H_N * H_N * 2);
  _Float16* Wft = (_Float16*)take(4ULL * NOUT * H_N * 2);
  float*    z   = (float*)   take((size_t)B_N * D_N * 4);
  _Float16* z16 = (_Float16*)take((size_t)B_N * D_N * 2);
  float*    h   = (float*)   take((size_t)B_N * H_N * 4);
  _Float16* a16 = (_Float16*)take((size_t)B_N * H_N * 2);
  _Float16* b16 = (_Float16*)take((size_t)B_N * H_N * 2);
  _Float16* h16 = (_Float16*)take((size_t)B_N * H_N * 2);
  float*    ldp = (float*)   take((size_t)B_N * 16 * 4);

  // ---- weight prep: mask + f16 + transpose ----
  for (int f = 0; f < 4; ++f)
    wconv_t<<<512, 256, 0, stream>>>(W0 + (size_t)f * D_N * H_N,
                                     W0t + (size_t)f * H_N * D_N, D_N, H_N, 0);
  for (int fi = 0; fi < 8; ++fi) {
    wconv_t<<<1024, 256, 0, stream>>>(Wa + (size_t)fi * H_N * H_N,
                                      Wat + (size_t)fi * H_N * H_N, H_N, H_N, 1);
    wconv_t<<<1024, 256, 0, stream>>>(Wb + (size_t)fi * H_N * H_N,
                                      Wbt + (size_t)fi * H_N * H_N, H_N, H_N, 1);
  }
  for (int f = 0; f < 4; ++f)
    wconv_t<<<2048, 256, 0, stream>>>(Wf + (size_t)f * H_N * NOUT,
                                      Wft + (size_t)f * NOUT * H_N, H_N, NOUT, 2);

  hipMemcpyAsync(z, x, (size_t)B_N * D_N * 4, hipMemcpyDeviceToDevice, stream);
  cvt16<<<2048, 256, 0, stream>>>(x, z16, B_N * D_N);

  dim3 ggrid(B_N / 128, H_N / 128);
  for (int f = 3; f >= 0; --f) {
    // h = z @ W0 + b0 ; a16 = relu(h)
    gemm_wmma<false, true, true, true><<<ggrid, 256, 0, stream>>>(
        z16, W0t + (size_t)f * H_N * D_N, b0 + f * H_N, nullptr, h, a16,
        B_N, H_N, D_N);
    for (int i = 0; i < 2; ++i) {
      int fi = f * 2 + i;
      // b16 = relu(a16 @ Wa + ba)
      gemm_wmma<false, false, true, true><<<ggrid, 256, 0, stream>>>(
          a16, Wat + (size_t)fi * H_N * H_N, ba + fi * H_N, nullptr, nullptr,
          b16, B_N, H_N, H_N);
      if (i == 0) {
        // h = h + b16 @ Wb + bb ; a16 = relu(h)
        gemm_wmma<true, true, true, true><<<ggrid, 256, 0, stream>>>(
            b16, Wbt + (size_t)fi * H_N * H_N, bb + fi * H_N, h, h, a16,
            B_N, H_N, H_N);
      } else {
        // last block: only h16 = f16(h + b16 @ Wb + bb) is needed downstream
        gemm_wmma<true, false, true, false><<<ggrid, 256, 0, stream>>>(
            b16, Wbt + (size_t)fi * H_N * H_N, bb + fi * H_N, h, nullptr, h16,
            B_N, H_N, H_N);
      }
    }
    gemm_spline<<<dim3(B_N / 16, 16), 256, 0, stream>>>(
        h16, Wft + (size_t)f * NOUT * H_N, bf + (size_t)f * NOUT, z, z16, ldp,
        (f != 3) ? 1 : 0);
  }

  finalize_kernel<<<B_N, 256, 0, stream>>>(z, ldp, (float*)d_out);
}
